// DeterministicModel_85048942396050
// MI455X (gfx1250) — matrix operand order repaired and checked by reading, not compiled, not run on hardware
//
#include <hip/hip_runtime.h>
#include <hip/hip_bf16.h>
#include <cstdint>
#include <cstddef>

typedef _Float16 v16h __attribute__((ext_vector_type(16)));
typedef _Float16 h8   __attribute__((ext_vector_type(8)));
typedef float    v8f  __attribute__((ext_vector_type(8)));

#define HH 224
#define WW 224
#define HWP (HH*WW)
#define FF 32
#define TT 8

// ---------------------------------------------------------------------------
// zero-fill helper (graph-capture safe; no hipMemset)
// ---------------------------------------------------------------------------
__global__ void zero_f32(float* __restrict__ p, int n) {
  int i = blockIdx.x * blockDim.x + threadIdx.x;
  if (i < n) p[i] = 0.f;
}

// ---------------------------------------------------------------------------
// Pack HWIO weights (Wx concat Wh along input channels) into wave32 WMMA
// B-fragment order for v_wmma_f32_16x16x32_f16:
//   packed[((kc*NT + nt)*32 + lane)*16 + h] = w(k = kc*32 + (h>>3)*16 + (lane>>4)*8 + (h&7),
//                                              n = nt*16 + (lane&15))
// where k = ((ky*ks)+kx)*cinT + c, zero-padded past kreal / ncolsReal.
// ---------------------------------------------------------------------------
__global__ void pack_weights(const float* __restrict__ Wx,
                             const float* __restrict__ Wh,
                             _Float16* __restrict__ out,
                             int ks, int cinX, int cinH, int ncolsReal,
                             int NT, int kchunks) {
  int idx = blockIdx.x * blockDim.x + threadIdx.x;
  int total = kchunks * NT * 512;
  if (idx >= total) return;
  int h  = idx & 15;
  int l  = (idx >> 4) & 31;
  int t  = idx >> 9;
  int nt = t % NT;
  int kc = t / NT;
  int cinT  = cinX + cinH;
  int kreal = ks * ks * cinT;
  int k = kc * 32 + ((h >> 3) << 4) + ((l >> 4) << 3) + (h & 7);
  int n = nt * 16 + (l & 15);
  float v = 0.f;
  if (k < kreal && n < ncolsReal) {
    int c  = k % cinT;
    int r  = k / cinT;
    int kx = r % ks;
    int ky = r / ks;
    if (c < cinX) v = Wx[((size_t)((ky*ks + kx)*cinX + c))*ncolsReal + n];
    else          v = Wh[((size_t)((ky*ks + kx)*cinH + (c - cinX)))*ncolsReal + n];
  }
  out[idx] = (_Float16)v;
}

__device__ __forceinline__ float hsig(float z) {
  return fminf(fmaxf(0.2f * z + 0.5f, 0.f), 1.f);
}

// ---------------------------------------------------------------------------
// Fused ConvLSTM step as implicit GEMM:
//   gates = conv([Xin, Hin], Wp) + bias        (one WMMA GEMM, K = ks*ks*cinT)
//   LSTM pointwise update in the epilogue (gate tiles live in the same wave).
// Block = 64 threads (2 waves), each wave owns 16 consecutive pixels of row y,
// block covers 32 pixels; grid = (224 rows, 7 x-slabs).
// Per chunk: 16 global_load_b128 (B fragments) are issued first, the LDS
// A-gather runs under them, then 8 WMMAs retire with partial loadcnt waits.
// Single B buffer keeps VGPRs ~150 so several waves/SIMD stay resident.
// FINAL=true instead does the 3x3x32->3 conv + sigmoid into Out3.
// ---------------------------------------------------------------------------
template<int NT, int KS, bool FINAL>
__global__ __launch_bounds__(64)
void convlstm_gemm(const float* __restrict__ Xin, int cinX,
                   const float* __restrict__ Hin,
                   const _Float16* __restrict__ Wp,
                   int kchunks, int kreal,
                   const float* __restrict__ bias,
                   float* __restrict__ Cst,
                   float* __restrict__ Hout,
                   const float* __restrict__ Resid, int resCin,
                   float* __restrict__ Xout,
                   float* __restrict__ Out3) {
  constexpr int PAD = KS / 2;
  constexpr int WID = 32 + 2 * PAD;
  const int y    = blockIdx.x;
  const int x0   = blockIdx.y * 32;
  const int tid  = threadIdx.x;
  const int wave = tid >> 5;
  const int lane = tid & 31;
  const int cinT = cinX + (Hin ? FF : 0);

  // f16 activation halo tile: [KS rows][WID cols][cinT channels]
  __shared__ __align__(16) _Float16 stage[5 * 36 * 64];  // 23 KB max

  const int total = KS * WID * cinT;
  for (int idx = tid; idx < total; idx += 64) {
    int c  = idx % cinT;
    int r  = idx / cinT;
    int xx = r % WID;
    int rr = r / WID;
    int iy = y  + rr - PAD;
    int ix = x0 + xx - PAD;
    float v = 0.f;
    if (iy >= 0 && iy < HH && ix >= 0 && ix < WW) {
      int p = iy * WW + ix;
      if (c < cinX) v = Xin[(size_t)p * cinX + c];
      else          v = Hin[(size_t)p * FF + (c - cinX)];
    }
    stage[idx] = (_Float16)v;
  }
  __syncthreads();

  v8f acc[NT];
#pragma unroll
  for (int nt = 0; nt < NT; ++nt)
#pragma unroll
    for (int e = 0; e < 8; ++e) acc[nt][e] = 0.f;

  const int m   = lane & 15;     // pixel within wave tile (A-matrix row M)
  const int pxl = wave * 16 + m; // local x (halo-relative: +kx gives LDS col)
  const v16h* __restrict__ Bl = reinterpret_cast<const v16h*>(Wp) + lane;

  for (int kc = 0; kc < kchunks; ++kc) {
    // ---- issue all B-fragment loads for this chunk first (16x b128)
    v16h b[NT];
#pragma unroll
    for (int nt = 0; nt < NT; ++nt)
      b[nt] = Bl[(size_t)(kc * NT + nt) * 32];

    if (kc + 1 < kchunks)  // warm L2 one chunk ahead
      __builtin_prefetch(&Bl[(size_t)(kc + 1) * NT * 32], 0, 0);

    // ---- A fragment from LDS (runs under the B loads)
    // ISA 16-bit A layout:
    //   lane<16 : halves 0..7 -> K=0..7,  halves 8..15 -> K=16..23
    //   lane>=16: halves 0..7 -> K=8..15, halves 8..15 -> K=24..31
    v16h a;
#pragma unroll
    for (int g = 0; g < 2; ++g) {
      int k0 = kc * 32 + g * 16 + ((lane >> 4) << 3);  // multiple of 8
      if ((cinT & 7) == 0 && k0 + 8 <= kreal) {
        // contiguous 8 halves in LDS -> 16B vector load (ds_load_b128)
        int c0 = k0 % cinT;
        int r  = k0 / cinT;
        int kx = r % KS;
        int ky = r / KS;
        h8 v = *reinterpret_cast<const h8*>(
            &stage[((size_t)(ky * WID + pxl + kx)) * cinT + c0]);
#pragma unroll
        for (int e = 0; e < 8; ++e) a[g * 8 + e] = v[e];
      } else {
#pragma unroll
        for (int e = 0; e < 8; ++e) {
          int k = k0 + e;
          _Float16 val = (_Float16)0.f;
          if (k < kreal) {
            int c  = k % cinT;
            int r  = k / cinT;
            int kx = r % KS;
            int ky = r / KS;
            val = stage[((size_t)(ky * WID + pxl + kx)) * cinT + c];
          }
          a[g * 8 + e] = val;
        }
      }
    }

    // ---- 8 WMMAs; compiler retires them with descending partial waits
#pragma unroll
    for (int nt = 0; nt < NT; ++nt)
      acc[nt] = __builtin_amdgcn_wmma_f32_16x16x32_f16(
          false, a, false, b[nt], (short)0, acc[nt], false, false);
  }

  // ---- epilogue (C layout: VGPR r, lane -> M = r + 8*(lane>=16), N = lane&15)
  if constexpr (FINAL) {
    const int n = lane & 15;
#pragma unroll
    for (int r = 0; r < 8; ++r) {
      int mm = r + ((lane >> 4) << 3);
      int p  = y * WW + x0 + wave * 16 + mm;
      if (n < 3) {
        float v = acc[0][r] + bias[n];
        Out3[(size_t)p * 3 + n] = 1.f / (1.f + expf(-v));
      }
    }
  } else {
    // gate channel fc in 0..31; gate tiles: i->acc[0/1], f->acc[2/3],
    // c->acc[4/5], o->acc[6/7] (sub-tile s = fc>=16)
#pragma unroll
    for (int r = 0; r < 8; ++r) {
      int mm = r + ((lane >> 4) << 3);
      int p  = y * WW + x0 + wave * 16 + mm;
#pragma unroll
      for (int s = 0; s < 2; ++s) {
        int fc = (lane & 15) + s * 16;
        float zi = acc[0 + s][r] + bias[fc];
        float zf = acc[2 + s][r] + bias[32 + fc];
        float zc = acc[4 + s][r] + bias[64 + fc];
        float zo = acc[6 + s][r] + bias[96 + fc];
        float ig = hsig(zi), fg = hsig(zf), og = hsig(zo);
        size_t sidx = (size_t)p * FF + fc;
        float cn = fg * Cst[sidx] + ig * tanhf(zc);
        Cst[sidx] = cn;
        float hh = og * tanhf(cn);
        Hout[sidx] = hh;
        if (Xout) {
          float res = 0.f;
          if (Resid && fc < resCin) res = Resid[(size_t)p * resCin + fc];
          Xout[sidx] = hh + res;
        }
      }
    }
  }
}

// ---------------------------------------------------------------------------
// Host orchestration (all launches on `stream`; graph-capture safe)
// ---------------------------------------------------------------------------
extern "C" void kernel_launch(void* const* d_in, const int* in_sizes, int n_in,
                              void* d_out, int out_size, void* d_ws, size_t ws_size,
                              hipStream_t stream) {
  (void)in_sizes; (void)n_in; (void)out_size; (void)ws_size;

  const float* x  = (const float*)d_in[0];
  const float* W0 = (const float*)d_in[1];
  const float* U0 = (const float*)d_in[2];
  const float* b0 = (const float*)d_in[3];
  const float* W1 = (const float*)d_in[4];
  const float* U1 = (const float*)d_in[5];
  const float* b1 = (const float*)d_in[6];
  const float* Wf = (const float*)d_in[7];
  const float* Uf = (const float*)d_in[8];
  const float* bf = (const float*)d_in[9];
  const float* Wc = (const float*)d_in[10];
  const float* bc = (const float*)d_in[11];
  float* out = (float*)d_out;

  const size_t HW = (size_t)HWP;
  char* wptr = (char*)d_ws;
  auto carve = [&](size_t bytes) -> char* {
    char* p = wptr;
    wptr += (bytes + 255) & ~(size_t)255;
    return p;
  };
  float* hA = (float*)carve(HW * FF * 4);            // h ping
  float* hB = (float*)carve(HW * FF * 4);            // h pong
  float* cS = (float*)carve(HW * FF * 4);            // c state
  float* XA = (float*)carve((size_t)TT * HW * FF * 4);  // layer0 out / Xfinal
  float* XB = (float*)carve((size_t)TT * HW * FF * 4);  // layer1 out

  const int KC0 = 28, KR0 = 5 * 5 * (3 + FF);   // 875 -> 28 chunks
  const int KC1 = 50, KR1 = 5 * 5 * (FF + FF);  // 1600 -> 50 chunks
  const int KCc = 9,  KRc = 3 * 3 * FF;         // 288 -> 9 chunks
  _Float16* P0 = (_Float16*)carve((size_t)KC0 * 8 * 512 * 2);
  _Float16* P1 = (_Float16*)carve((size_t)KC1 * 8 * 512 * 2);
  _Float16* Pf = (_Float16*)carve((size_t)KC1 * 8 * 512 * 2);
  _Float16* Pc = (_Float16*)carve((size_t)KCc * 1 * 512 * 2);

  auto pack = [&](const float* wx, const float* wh, _Float16* dst,
                  int ks, int cx, int ch, int ncr, int nt, int kch) {
    int total = kch * nt * 512;
    pack_weights<<<dim3((total + 255) / 256), dim3(256), 0, stream>>>(
        wx, wh, dst, ks, cx, ch, ncr, nt, kch);
  };
  pack(W0, U0, P0, 5, 3,  32, 128, 8, KC0);
  pack(W1, U1, P1, 5, 32, 32, 128, 8, KC1);
  pack(Wf, Uf, Pf, 5, 32, 32, 128, 8, KC1);
  pack(Wc, nullptr, Pc, 3, 32, 0, 3, 1, KCc);

  auto zero = [&](float* p, size_t n) {
    zero_f32<<<dim3((unsigned)((n + 255) / 256)), dim3(256), 0, stream>>>(p, (int)n);
  };

  dim3 grid(HH, 7), block(64);

  // ---- layer 0: x (3ch) -> XA = h + pad(x)
  zero(hA, HW * FF); zero(cS, HW * FF);
  for (int t = 0; t < TT; ++t) {
    const float* hin = (t & 1) ? hB : hA;
    float* hout      = (t & 1) ? hA : hB;
    convlstm_gemm<8, 5, false><<<grid, block, 0, stream>>>(
        x + (size_t)t * HW * 3, 3, hin, P0, KC0, KR0, b0, cS, hout,
        x + (size_t)t * HW * 3, 3, XA + (size_t)t * HW * FF, nullptr);
  }

  // ---- layer 1: XA -> XB = h + XA
  zero(hA, HW * FF); zero(cS, HW * FF);
  for (int t = 0; t < TT; ++t) {
    const float* hin = (t & 1) ? hB : hA;
    float* hout      = (t & 1) ? hA : hB;
    convlstm_gemm<8, 5, false><<<grid, block, 0, stream>>>(
        XA + (size_t)t * HW * FF, FF, hin, P1, KC1, KR1, b1, cS, hout,
        XA + (size_t)t * HW * FF, FF, XB + (size_t)t * HW * FF, nullptr);
  }

  // ---- layer f: XB -> (at t=T-1) Xfinal = h_T + pad(x[:, -1])  (reuse XA)
  zero(hA, HW * FF); zero(cS, HW * FF);
  for (int t = 0; t < TT; ++t) {
    const float* hin = (t & 1) ? hB : hA;
    float* hout      = (t & 1) ? hA : hB;
    bool last = (t == TT - 1);
    convlstm_gemm<8, 5, false><<<grid, block, 0, stream>>>(
        XB + (size_t)t * HW * FF, FF, hin, Pf, KC1, KR1, bf, cS, hout,
        last ? (x + (size_t)(TT - 1) * HW * 3) : nullptr, 3,
        last ? XA : nullptr, nullptr);
  }

  // ---- final 3x3 conv -> sigmoid -> d_out [224,224,3]
  convlstm_gemm<1, 3, true><<<grid, block, 0, stream>>>(
      XA, FF, nullptr, Pc, KCc, KRc, bc, nullptr, nullptr,
      nullptr, 0, nullptr, out);
}
